// MTGRU_44212393345725
// MI455X (gfx1250) — compile-verified, matrix operand order checked by reading, not run on hardware
//
#include <hip/hip_runtime.h>
#include <hip/hip_bf16.h>
#include <math.h>

// ---------------- CDNA5 WMMA types & helpers ----------------
typedef __bf16  bf16x16 __attribute__((ext_vector_type(16)));
typedef float   f32x8   __attribute__((ext_vector_type(8)));

#define DEV __device__ __forceinline__

DEV f32x8 wmma_bf16(bf16x16 a, bf16x16 b, f32x8 c) {
    // D = A(16x32 bf16) * B(32x16 bf16) + C(16x16 f32)
    return __builtin_amdgcn_wmma_f32_16x16x32_bf16(false, a, false, b,
                                                   (short)0, c, false, false);
}

// ---- WMMA fragment storage: one 32x16(B) or 16x32(A) tile = 32 lanes x 16 bf16,
// ---- contiguous (1 KB). Fragment load = 2x global_load_b128 per lane.
DEV bf16x16 load_frag(const __bf16* F, int tile) {
    const uint4* p = (const uint4*)(F + ((size_t)tile << 9) + (size_t)(threadIdx.x & 31) * 16);
    union { uint4 u[2]; bf16x16 v; } t;
    t.u[0] = p[0];
    t.u[1] = p[1];
    return t.v;
}

// Per-lane K-packing (ISA 7.12.2): lanes 0-15 hold K={0..7,16..23}, lanes 16-31 hold K={8..15,24..31}.
DEV int frag_koff(int lane, int e) {
    return (e & 7) + ((e & 8) ? 16 : 0) + ((lane & 16) ? 8 : 0);
}

// Scatter one element (m,k) of an A-matrix [M x K] into A-fragment layout (KC = K/32 chunks).
DEV void scatter_afrag(__bf16* F, int KC, int m, int k, float v) {
    const int kc = k >> 5, kk = k & 31;
    const int lanehi = (kk >> 3) & 1;
    const int e  = (kk & 7) + ((kk & 16) ? 8 : 0);
    const int ln = (lanehi << 4) | (m & 15);
    F[((size_t)((m >> 4) * KC + kc) << 9) + ln * 16 + e] = (__bf16)v;
}

// Inverse mapping for B-fragments: (tile, r) -> (k, n), NT = N/16 tiles.
DEV void frag_kn_b(int tile, int r, int NT, int& k, int& n) {
    const int lane = r >> 4, e = r & 15;
    const int kc = tile / NT, nt = tile - kc * NT;
    n = nt * 16 + (lane & 15);
    k = kc * 32 + frag_koff(lane, e);
}

// A-tile loader from bf16 row-major matrix (lda, k0 multiples of 8/32 -> 16B aligned halves).
DEV bf16x16 load_a_bf16(const __bf16* A, int lda, int row0, int k0) {
    const int lane = threadIdx.x & 31;
    const int m  = row0 + (lane & 15);
    const int kb = k0 + ((lane & 16) ? 8 : 0);
    const __bf16* p = A + (size_t)m * lda + kb;
    union { uint4 u[2]; bf16x16 v; } t;
    t.u[0] = *(const uint4*)p;
    t.u[1] = *(const uint4*)(p + 16);
    return t.v;
}

// B-tile (32x16) loader from row-major [K x N] fp32 matrix (convert->bf16 on the fly).
DEV bf16x16 load_b_f32(const float* B, size_t ldb, int k0, size_t col0) {
    const int lane = threadIdx.x & 31;
    const size_t n = col0 + (lane & 15);
    const int kb   = k0 + ((lane & 16) ? 8 : 0);
    bf16x16 b;
#pragma unroll
    for (int i = 0; i < 8; ++i) b[i]     = (__bf16)B[(size_t)(kb + i) * ldb + n];
#pragma unroll
    for (int i = 0; i < 8; ++i) b[8 + i] = (__bf16)B[(size_t)(kb + 16 + i) * ldb + n];
    return b;
}

// C/D layout: VGPR r -> M=row0+r (lanes 0-15) / row0+8+r (lanes 16-31); N=col0+(lane&15).
DEV void store_c_f32(float* C, int ldc, int row0, int col0, const f32x8& acc) {
    const int lane = threadIdx.x & 31;
    const int n    = col0 + (lane & 15);
    const int mo   = (lane & 16) ? 8 : 0;
#pragma unroll
    for (int r = 0; r < 8; ++r) C[(size_t)(row0 + mo + r) * ldc + n] = acc[r];
}

DEV float sigmoidf_(float x) { return 1.0f / (1.0f + expf(-x)); }

// ---------------- Model dimensions ----------------
#define Bn   64
#define Tn   128
#define DIN  32
#define DOUT 64
#define HE   512
#define HD   1024
#define Kz   12
#define NPH  512
#define NPL  64
#define NPSI 1148928u
#define OFF_BH   1048576u
#define OFF_WIH  1049600u
#define OFF_C    1082368u
#define OFF_D    1147904u
// fragment-grid geometry
#define ENC_KC 18   /* 576/32  */
#define ENC_NT 96   /* 1536/16 */
#define GRU_KC 33   /* 1056/32 */
#define GRU_NT 128  /* 2048/16 */
#define HENC_KC 16  /* 512/32  */
#define HDEC_KC 32  /* 1024/32 */

// ============ K0: fragment packing (bf16) + state init ============
__global__ void mtgru_pack(const float* __restrict__ enc_Wih, const float* __restrict__ enc_Whh,
                           const float* __restrict__ gru_Wih, const float* __restrict__ gru_Whh,
                           const float* __restrict__ outputs, const float* __restrict__ inputs,
                           const float* __restrict__ state,
                           __bf16* __restrict__ encWf, __bf16* __restrict__ gruWf,
                           __bf16* __restrict__ encXf, __bf16* __restrict__ decXf,
                           __bf16* __restrict__ hencF, __bf16* __restrict__ hdecF,
                           float* __restrict__ henc, float* __restrict__ hdec) {
    const int idx = blockIdx.x * blockDim.x + threadIdx.x;
    const int stride = gridDim.x * blockDim.x;
    // encWf: B-frag of [576 x 1536] = concat_k( enc_Wih^T, enc_Whh^T )
    for (int i = idx; i < ENC_KC * ENC_NT * 512; i += stride) {
        int k, n; frag_kn_b(i >> 9, i & 511, ENC_NT, k, n);
        const float v = (k < DOUT) ? enc_Wih[n * DOUT + k] : enc_Whh[n * HE + (k - DOUT)];
        encWf[i] = (__bf16)v;
    }
    // gruWf: B-frag of [1056 x 2048] = concat_k( gru_Wih, gru_Whh )
    for (int i = idx; i < GRU_KC * GRU_NT * 512; i += stride) {
        int k, n; frag_kn_b(i >> 9, i & 511, GRU_NT, k, n);
        const float v = (k < DIN) ? gru_Wih[k * 2048 + n] : gru_Whh[(k - DIN) * 2048 + n];
        gruWf[i] = (__bf16)v;
    }
    // encXf: per-t A-frags of outputs[:,t,:] [64 x 64] -> 8 tiles/t
    for (int i = idx; i < Tn * 8 * 512; i += stride) {
        const int t = i >> 12, r2 = i & 4095, tile = r2 >> 9, r = r2 & 511;
        const int mt = tile >> 1, kc = tile & 1, lane = r >> 4, e = r & 15;
        const int m = mt * 16 + (lane & 15);
        const int k = kc * 32 + frag_koff(lane, e);
        encXf[i] = (__bf16)outputs[(size_t)m * (Tn * DOUT) + t * DOUT + k];
    }
    // decXf: per-t A-frags of inputs[:,t,:] [64 x 32] -> 4 tiles/t
    for (int i = idx; i < Tn * 4 * 512; i += stride) {
        const int t = i >> 11, r2 = i & 2047, tile = r2 >> 9, r = r2 & 511;
        const int lane = r >> 4, e = r & 15;
        const int m = tile * 16 + (lane & 15);
        const int k = frag_koff(lane, e);
        decXf[i] = (__bf16)inputs[(size_t)m * (Tn * DIN) + t * DIN + k];
    }
    // hencF: zero (h0 = 0); hdecF from initial state
    for (int i = idx; i < 4 * HENC_KC * 512; i += stride) hencF[i] = (__bf16)0.0f;
    for (int i = idx; i < 4 * HDEC_KC * 512; i += stride) {
        const int tile = i >> 9, r = i & 511;
        const int mt = tile >> 5, kc = tile & 31, lane = r >> 4, e = r & 15;
        const int m = mt * 16 + (lane & 15);
        const int k = kc * 32 + frag_koff(lane, e);
        hdecF[i] = (__bf16)state[m * HD + k];
    }
    for (int i = idx; i < Bn * HE; i += stride) henc[i] = 0.0f;
    for (int i = idx; i < Bn * HD; i += stride) hdec[i] = state[i];
}

// ============ K1: encoder step GEMM (pure fragment WMMA) ============
__global__ void mtgru_enc_gemm(const __bf16* __restrict__ encXf, const __bf16* __restrict__ hencF,
                               const __bf16* __restrict__ encWf,
                               float* __restrict__ Gi, float* __restrict__ Gh, int t) {
    const int wave = (blockIdx.x * blockDim.x + threadIdx.x) >> 5;   // 384 waves = 4 x 96 tiles
    const int mt = wave & 3, nt = wave >> 2;
    f32x8 ai = {}, ah = {};
    const __bf16* Ax = encXf + (size_t)t * 4096;
#pragma unroll
    for (int c = 0; c < 2; ++c) {                                    // K = 0..63 (input part)
        ai = wmma_bf16(load_frag(Ax, mt * 2 + c), load_frag(encWf, c * ENC_NT + nt), ai);
    }
#pragma unroll 4
    for (int c = 0; c < 16; ++c) {                                   // K = 64..575 (hidden part)
        ah = wmma_bf16(load_frag(hencF, mt * HENC_KC + c),
                       load_frag(encWf, (2 + c) * ENC_NT + nt), ah);
    }
    store_c_f32(Gi, 1536, mt * 16, nt * 16, ai);
    store_c_f32(Gh, 1536, mt * 16, nt * 16, ah);
}

// ============ K2: encoder GRU gate fusion (also refreshes h A-fragments) ============
__global__ void mtgru_enc_gate(const float* __restrict__ Gi, const float* __restrict__ Gh,
                               const float* __restrict__ bih, const float* __restrict__ bhh,
                               float* __restrict__ henc, __bf16* __restrict__ hencF) {
    const int i = blockIdx.x * blockDim.x + threadIdx.x;   // 64*512 threads
    const int b = i >> 9, n = i & 511;
    const float* gi = Gi + b * 1536;
    const float* gh = Gh + b * 1536;
    const float r  = sigmoidf_(gi[n] + bih[n] + gh[n] + bhh[n]);
    const float z  = sigmoidf_(gi[HE + n] + bih[HE + n] + gh[HE + n] + bhh[HE + n]);
    const float nn = tanhf(gi[2 * HE + n] + bih[2 * HE + n] + r * (gh[2 * HE + n] + bhh[2 * HE + n]));
    const float hn = (1.0f - z) * nn + z * henc[i];
    henc[i] = hn;
    scatter_afrag(hencF, HENC_KC, b, n, hn);
}

// ============ K3: latent head + hypernet layers 1&2 (per-sample workgroup) ============
__global__ void mtgru_latent(const float* __restrict__ henc, const float* __restrict__ eps,
                             const float* __restrict__ to_mu, const float* __restrict__ to_ls,
                             const float* __restrict__ ls_bias,
                             const float* __restrict__ W1, const float* __restrict__ b1,
                             const float* __restrict__ W2, const float* __restrict__ b2,
                             float* __restrict__ out_mu, float* __restrict__ out_ls,
                             __bf16* __restrict__ a2f) {
    __shared__ float hl[HE];
    __shared__ float zl[Kz];
    __shared__ float a1[NPH];
    const int b = blockIdx.x, tid = threadIdx.x;          // 512 threads
    hl[tid] = henc[b * HE + tid];
    __syncthreads();
    if (tid < Kz) {
        float m = 0.0f, s = 0.0f;
        for (int j = 0; j < HE; ++j) { m += hl[j] * to_mu[j * Kz + tid]; s += hl[j] * to_ls[j * Kz + tid]; }
        s += ls_bias[tid];
        out_mu[b * Kz + tid] = m;
        out_ls[b * Kz + tid] = s;
        zl[tid] = m + eps[b * Kz + tid] * expf(s);
    }
    __syncthreads();
    {
        float acc = b1[tid];
#pragma unroll
        for (int l = 0; l < Kz; ++l) acc += zl[l] * W1[l * NPH + tid];
        a1[tid] = tanhf(acc);
    }
    __syncthreads();
    if (tid < NPL) {
        float acc = b2[tid];
        for (int j = 0; j < NPH; ++j) acc += a1[j] * W2[j * NPL + tid];
        // write a2 [64 x 64] directly as A-fragments (2 k-chunks x 4 m-tiles)
        scatter_afrag(a2f, 2, b, tid, acc);
    }
}

// ============ K4: hypernet layer 3: psi = a2 @ W3 + b3, scatter into per-sample params ============
__global__ void mtgru_psi(const __bf16* __restrict__ a2f, const float* __restrict__ W3,
                          const float* __restrict__ b3,
                          __bf16* __restrict__ Whh, float* __restrict__ bh,
                          __bf16* __restrict__ Wih, __bf16* __restrict__ Cb,
                          __bf16* __restrict__ Db) {
    const int wave = (blockIdx.x * blockDim.x + threadIdx.x) >> 5;    // 287232 waves
    const int mt = wave & 3;
    const size_t nt = (size_t)(wave >> 2);                            // 0..71807
    const int row0 = mt * 16;
    const size_t col0 = nt * 16;
    const int lane = threadIdx.x & 31;
    // streaming prefetch of the 294 MB W3 matrix ahead of this wave (global_prefetch_b8)
    __builtin_prefetch(W3 + (size_t)(lane & 15) * NPSI + col0 + 4096, 0, 0);
    f32x8 acc = {};
#pragma unroll
    for (int c = 0; c < 2; ++c) {                                     // K = 64
        acc = wmma_bf16(load_frag(a2f, mt * 2 + c),
                        load_b_f32(W3, (size_t)NPSI, c * 32, col0), acc);
    }
    // Epilogue: add bias, convert & scatter to per-sample parameter tensors.
    const size_t j  = col0 + (size_t)(lane & 15);
    const int    mo = (lane & 16) ? 8 : 0;
    const float  bias = b3[j];
#pragma unroll
    for (int r = 0; r < 8; ++r) {
        const int brow = row0 + mo + r;
        const float v  = acc[r] + bias;
        if (j < OFF_BH) {
            Whh[(size_t)brow * (HD * HD) + j] = (__bf16)v;            // [h*HD + g] row-major
        } else if (j < OFF_WIH) {
            bh[brow * HD + (int)(j - OFF_BH)] = v;                    // fp32 bias
        } else if (j < OFF_C) {
            Wih[(size_t)brow * (DIN * HD) + (j - OFF_WIH)] = (__bf16)v;
        } else if (j < OFF_D) {
            // C_b stored directly in B-fragment order ([1024 x 64], KC=32, NT=4)
            const unsigned local = (unsigned)(j - OFF_C);
            const int k = (int)(local >> 6), n = (int)(local & 63);
            const int kc = k >> 5, kk = k & 31;
            const int lanehi = (kk >> 3) & 1;
            const int e  = (kk & 7) + ((kk & 16) ? 8 : 0);
            const int ln = (lanehi << 4) | (n & 15);
            Cb[(size_t)brow * (HD * DOUT)
               + ((size_t)(kc * 4 + (n >> 4)) << 9) + ln * 16 + e] = (__bf16)v;
        } else {
            Db[brow * (DIN * DIN) + (int)(j - OFF_D)] = (__bf16)v;
        }
    }
}

// ============ K5: decoder step shared-gate GEMM (pure fragment WMMA) ============
__global__ void mtgru_dec_gemm(const __bf16* __restrict__ decXf, const __bf16* __restrict__ hdecF,
                               const __bf16* __restrict__ gruWf, float* __restrict__ G, int t) {
    const int wave = (blockIdx.x * blockDim.x + threadIdx.x) >> 5;    // 512 waves = 4 x 128 tiles
    const int mt = wave & 3, nt = wave >> 2;
    f32x8 acc = {};
    const __bf16* Ax = decXf + (size_t)t * 2048;
    acc = wmma_bf16(load_frag(Ax, mt), load_frag(gruWf, nt), acc);    // K-chunk 0: x_t
#pragma unroll 4
    for (int c = 0; c < 32; ++c) {                                    // K-chunks 1..32: h
        acc = wmma_bf16(load_frag(hdecF, mt * HDEC_KC + c),
                        load_frag(gruWf, (c + 1) * GRU_NT + nt), acc);
    }
    store_c_f32(G, 2048, mt * 16, nt * 16, acc);
}

// ============ K6: decoder gates + per-sample bf16 matvec (L2-resident Whh_b) ============
__global__ void __launch_bounds__(1024)
mtgru_dec_gate(const float* __restrict__ inputs, const float* __restrict__ G,
               const float* __restrict__ gru_bias,
               const __bf16* __restrict__ Whh, const float* __restrict__ bh,
               const __bf16* __restrict__ Wih,
               float* __restrict__ hdec, __bf16* __restrict__ hdecF,
               __bf16* __restrict__ decbf, float* __restrict__ states, int t) {
    __shared__ float hs[HD];
    __shared__ float xs[DIN];
    const int b = blockIdx.x, g = threadIdx.x;            // 64 blocks x 1024 threads
    hs[g] = hdec[b * HD + g];
    if (g < DIN) xs[g] = inputs[(size_t)b * Tn * DIN + t * DIN + g];
    __syncthreads();
    const float r  = sigmoidf_(G[b * 2048 + g] + gru_bias[g]);
    const float zt = sigmoidf_(G[b * 2048 + HD + g] + gru_bias[HD + g]);
    const __bf16* w = Whh + (size_t)b * (HD * HD) + g;    // stride HD over h -> coalesced across g
    float mv = 0.0f;
#pragma unroll 8
    for (int h = 0; h < HD; ++h) mv += hs[h] * (float)w[(size_t)h * HD];
    const __bf16* wi = Wih + (size_t)b * (DIN * HD) + g;
    float xv = 0.0f;
#pragma unroll
    for (int i = 0; i < DIN; ++i) xv += xs[i] * (float)wi[(size_t)i * HD];
    const float eta = tanhf(xv + r * mv + bh[b * HD + g]);
    const float hn  = zt * hs[g] + (1.0f - zt) * eta;
    hdec[b * HD + g] = hn;
    scatter_afrag(hdecF, HDEC_KC, b, g, hn);
    decbf[(size_t)b * Tn * HD + (size_t)t * HD + g] = (__bf16)hn;
    if (t == Tn - 1) states[b * HD + g] = hn;
}

// ============ K7: output projection yhat = dec@C + resid (per-sample WMMA) ============
__global__ void mtgru_out(const __bf16* __restrict__ decbf, const __bf16* __restrict__ Cb,
                          const __bf16* __restrict__ Db, const float* __restrict__ inputs,
                          float* __restrict__ yhat) {
    const int wave = (blockIdx.x * blockDim.x + threadIdx.x) >> 5;    // 2048 waves
    const int b  = wave >> 5;
    const int tl = wave & 31;
    const int mt = tl >> 2, nt = tl & 3;
    const int row0 = mt * 16, col0 = nt * 16;
    const __bf16* A  = decbf + (size_t)b * Tn * HD;                   // [128 x 1024] bf16 row-major
    const __bf16* Bf = Cb + (size_t)b * (HD * DOUT);                  // B-fragments, KC=32, NT=4
    f32x8 acc = {};
#pragma unroll 4
    for (int c = 0; c < 32; ++c) {
        acc = wmma_bf16(load_a_bf16(A, HD, row0, c * 32), load_frag(Bf, c * 4 + nt), acc);
    }
    const int lane = threadIdx.x & 31;
    const int o  = col0 + (lane & 15);
    const int mo = (lane & 16) ? 8 : 0;
    const float*  xb = inputs + (size_t)b * Tn * DIN;
    const __bf16* Dp = Db + (size_t)b * (DIN * DIN);
#pragma unroll
    for (int r = 0; r < 8; ++r) {
        const int tt = row0 + mo + r;
        float v = acc[r];
        if (o < DIN) {
            v += xb[tt * DIN + o];
        } else {
            float d = 0.0f;
#pragma unroll
            for (int i = 0; i < DIN; ++i) d += xb[tt * DIN + i] * (float)Dp[i * DIN + (o - DIN)];
            v += d;
        }
        yhat[((size_t)b * Tn + tt) * DOUT + o] = v;
    }
}

// ============================ host launcher ============================
extern "C" void kernel_launch(void* const* d_in, const int* in_sizes, int n_in,
                              void* d_out, int out_size, void* d_ws, size_t ws_size,
                              hipStream_t stream) {
    (void)in_sizes; (void)n_in; (void)out_size; (void)ws_size;
    const float* inputs   = (const float*)d_in[0];
    const float* outputs  = (const float*)d_in[1];
    const float* eps      = (const float*)d_in[2];
    const float* state    = (const float*)d_in[3];
    const float* enc_Wih  = (const float*)d_in[4];
    const float* enc_Whh  = (const float*)d_in[5];
    const float* enc_bih  = (const float*)d_in[6];
    const float* enc_bhh  = (const float*)d_in[7];
    const float* to_mu    = (const float*)d_in[8];
    const float* to_ls    = (const float*)d_in[9];
    const float* ls_bias  = (const float*)d_in[10];
    const float* psi_W1   = (const float*)d_in[11];
    const float* psi_b1   = (const float*)d_in[12];
    const float* psi_W2   = (const float*)d_in[13];
    const float* psi_b2   = (const float*)d_in[14];
    const float* psi_W3   = (const float*)d_in[15];
    const float* psi_b3   = (const float*)d_in[16];
    const float* gru_Wih  = (const float*)d_in[17];
    const float* gru_Whh  = (const float*)d_in[18];
    const float* gru_bias = (const float*)d_in[19];

    // Output layout: yhats [B,T,DOUT] | mu [B,K] | logstd [B,K] | states [B,HD]
    float* o_yhat   = (float*)d_out;
    float* o_mu     = o_yhat + (size_t)Bn * Tn * DOUT;
    float* o_ls     = o_mu + Bn * Kz;
    float* o_states = o_ls + Bn * Kz;

    // Workspace carve
    char* w = (char*)d_ws;
    size_t off = 0;
    auto take = [&](size_t bytes) { void* p = w + off; off += (bytes + 255) & ~(size_t)255; return p; };
    __bf16* encWf = (__bf16*)take((size_t)ENC_KC * ENC_NT * 512 * 2);
    __bf16* gruWf = (__bf16*)take((size_t)GRU_KC * GRU_NT * 512 * 2);
    __bf16* encXf = (__bf16*)take((size_t)Tn * 8 * 512 * 2);
    __bf16* decXf = (__bf16*)take((size_t)Tn * 4 * 512 * 2);
    __bf16* hencF = (__bf16*)take((size_t)4 * HENC_KC * 512 * 2);
    __bf16* hdecF = (__bf16*)take((size_t)4 * HDEC_KC * 512 * 2);
    float*  henc  = (float*) take((size_t)Bn * HE * 4);
    float*  hdec  = (float*) take((size_t)Bn * HD * 4);
    float*  Gi    = (float*) take((size_t)Bn * 1536 * 4);
    float*  Gh    = (float*) take((size_t)Bn * 1536 * 4);
    __bf16* a2f   = (__bf16*)take((size_t)Bn * NPL * 2);
    float*  Gdec  = (float*) take((size_t)Bn * 2048 * 4);
    __bf16* Whh   = (__bf16*)take((size_t)Bn * HD * HD * 2);      // 128 MB -> L2-resident
    float*  bh    = (float*) take((size_t)Bn * HD * 4);
    __bf16* Wih   = (__bf16*)take((size_t)Bn * DIN * HD * 2);
    __bf16* Cb    = (__bf16*)take((size_t)Bn * HD * DOUT * 2);
    __bf16* Db    = (__bf16*)take((size_t)Bn * DIN * DIN * 2);
    __bf16* decbf = (__bf16*)take((size_t)Bn * Tn * HD * 2);

    // 0) pack everything into WMMA fragment order, init states
    mtgru_pack<<<2048, 256, 0, stream>>>(enc_Wih, enc_Whh, gru_Wih, gru_Whh,
                                         outputs, inputs, state,
                                         encWf, gruWf, encXf, decXf, hencF, hdecF,
                                         henc, hdec);
    // 1) encoder GRU over time
    for (int t = 0; t < Tn; ++t) {
        mtgru_enc_gemm<<<96, 128, 0, stream>>>(encXf, hencF, encWf, Gi, Gh, t);
        mtgru_enc_gate<<<128, 256, 0, stream>>>(Gi, Gh, enc_bih, enc_bhh, henc, hencF);
    }
    // 2) latent head + hypernet layers 1&2
    mtgru_latent<<<64, 512, 0, stream>>>(henc, eps, to_mu, to_ls, ls_bias,
                                         psi_W1, psi_b1, psi_W2, psi_b2,
                                         o_mu, o_ls, a2f);
    // 3) hypernet layer 3 (streams 294 MB of psi_W3; WMMA + scatter epilogue)
    mtgru_psi<<<35904, 256, 0, stream>>>(a2f, psi_W3, psi_b3, Whh, bh, Wih, Cb, Db);
    // 4) decoder GRU over time
    for (int t = 0; t < Tn; ++t) {
        mtgru_dec_gemm<<<128, 128, 0, stream>>>(decXf, hdecF, gruWf, Gdec, t);
        mtgru_dec_gate<<<64, 1024, 0, stream>>>(inputs, Gdec, gru_bias, Whh, bh, Wih,
                                                hdec, hdecF, decbf, o_states, t);
    }
    // 5) output projection + residual
    mtgru_out<<<256, 256, 0, stream>>>(decbf, Cb, Db, inputs, o_yhat);
}
